// AttentionPooling_24378234372356
// MI455X (gfx1250) — matrix operand order repaired and checked by reading, not compiled, not run on hardware
//
#include <hip/hip_runtime.h>
#include <stdint.h>

typedef __attribute__((ext_vector_type(16))) __bf16 v16bf;
typedef __attribute__((ext_vector_type(8)))  float  v8f;

union BFPack { v16bf v; uint32_t u[8]; };

#define BB 128
#define SS 2048
#define DD 128
#define NQ 18
#define SC 64               // S-chunk per iteration
#define NCHUNK (SS / SC)

// single packed f32x2 -> bf16x2 convert (RNE); backend has this opcode
__device__ __forceinline__ uint32_t bf16_2(float lo, float hi) {
    uint32_t r;
    asm("v_cvt_pk_bf16_f32 %0, %1, %2" : "=v"(r) : "v"(lo), "v"(hi));
    return r;
}

__global__ __launch_bounds__(256) void attn_pool_kernel(
    const float* __restrict__ x, const float* __restrict__ q_emb,
    const int* __restrict__ questions, const int* __restrict__ mask,
    float* __restrict__ out)
{
    // ~62.4 KB static LDS
    __shared__ uint32_t Qs32[32 * 64];          // Q bf16 pairs, row-major (32 x 128)
    __shared__ uint32_t Xrow32[SC * 64];        // x chunk bf16, s-major  (QK B-operand)
    __shared__ uint32_t Xt32[DD * (SC / 2)];    // x chunk bf16, d-major  (AV B-operand)
    __shared__ float    Sbuf[32 * SC];          // fp32 scores (+bias)
    __shared__ uint32_t Pbuf32[32 * (SC / 2)];  // bf16 probabilities
    __shared__ float    Mbias[SS];              // mask -> additive bias, whole row
    __shared__ float    redbuf[256];
    __shared__ float    redbuf2[256];
    __shared__ float    m_state[32], l_state[32], rescale[32];

    const int b    = blockIdx.x;
    const int tid  = threadIdx.x;
    const int lane = tid & 31;
    const int wave = tid >> 5;
    const int lm   = lane & 15;
    const int h    = lane >> 4;        // half-wave select
    const int mt   = wave >> 2;        // M-tile (0..1)
    const int qq   = wave & 3;         // QK: s-tile; AV: d-tile base
    const float scale   = 0.08838834764831845f;  // 1/sqrt(128)
    const float NEG_INF = -__builtin_inff();

    if (tid < 32) { m_state[tid] = NEG_INF; l_state[tid] = 0.f; rescale[tid] = 0.f; }

    // ---- mask -> bias table (once per block) ----
    {
        #pragma unroll
        for (int j = 0; j < SS / 256; ++j) {
            int s = j * 256 + tid;
            Mbias[s] = (mask[(size_t)b * SS + s] == 1) ? 0.f : NEG_INF;
        }
    }

    // ---- Gather Q -> LDS bf16; rows 18..31 zero ----
    {
        int row  = tid >> 3;
        int dsg  = (tid & 7) << 4;
        if (row < NQ) {
            int qi = questions[b * NQ + row];
            const float* src = q_emb + (size_t)qi * DD + dsg;
            #pragma unroll
            for (int j = 0; j < 8; ++j)
                Qs32[row * 64 + (dsg >> 1) + j] = bf16_2(src[2 * j], src[2 * j + 1]);
        } else {
            #pragma unroll
            for (int j = 0; j < 8; ++j) Qs32[row * 64 + (dsg >> 1) + j] = 0u;
        }
    }

    // ---- x-tile prefetch state (software pipeline) ----
    const int sp   = tid >> 3;            // s-pair 0..31
    const int dseg = (tid & 7) << 4;      // d segment 0..112
    float4 pf[8];
    {   // prologue: issue loads for chunk 0
        const float4* r0 = (const float4*)(x + ((size_t)b * SS + 2 * sp) * DD + dseg);
        const float4* r1 = (const float4*)(x + ((size_t)b * SS + 2 * sp + 1) * DD + dseg);
        pf[0] = r0[0]; pf[1] = r0[1]; pf[2] = r0[2]; pf[3] = r0[3];
        pf[4] = r1[0]; pf[5] = r1[1]; pf[6] = r1[2]; pf[7] = r1[3];
    }

    __syncthreads();

    // ---- Build this wave's Q A-operands (16-bit A 16x32 layout), reused all chunks ----
    BFPack qa[4];
    #pragma unroll
    for (int kk = 0; kk < 4; ++kk) {
        #pragma unroll
        for (int v = 0; v < 8; ++v) {
            int idx = 16 * kk + (v < 4 ? 4 * h + v : 8 + 4 * h + (v - 4));
            qa[kk].u[v] = Qs32[(mt * 16 + lm) * 64 + idx];
        }
    }

    v8f accO[2];
    #pragma unroll
    for (int r = 0; r < 8; ++r) { accO[0][r] = 0.f; accO[1][r] = 0.f; }

    for (int ch = 0; ch < NCHUNK; ++ch) {
        const int s0 = ch * SC;

        // ---- Phase A: convert prefetched registers -> LDS (both layouts) ----
        {
            float a0[16], a1[16];
            #pragma unroll
            for (int j = 0; j < 4; ++j) {
                a0[4*j] = pf[j].x;   a0[4*j+1] = pf[j].y;   a0[4*j+2] = pf[j].z;   a0[4*j+3] = pf[j].w;
                a1[4*j] = pf[4+j].x; a1[4*j+1] = pf[4+j].y; a1[4*j+2] = pf[4+j].z; a1[4*j+3] = pf[4+j].w;
            }
            #pragma unroll
            for (int j = 0; j < 8; ++j) {
                Xrow32[(2 * sp    ) * 64 + (dseg >> 1) + j] = bf16_2(a0[2*j], a0[2*j+1]);
                Xrow32[(2 * sp + 1) * 64 + (dseg >> 1) + j] = bf16_2(a1[2*j], a1[2*j+1]);
            }
            #pragma unroll
            for (int j = 0; j < 16; ++j)
                Xt32[(dseg + j) * (SC / 2) + sp] = bf16_2(a0[j], a1[j]);
        }
        __syncthreads();

        // ---- issue prefetch for next chunk (overlaps phases B..E) ----
        if (ch + 1 < NCHUNK) {
            const float4* r0 = (const float4*)(x + ((size_t)b * SS + (s0 + SC) + 2 * sp) * DD + dseg);
            const float4* r1 = (const float4*)(x + ((size_t)b * SS + (s0 + SC) + 2 * sp + 1) * DD + dseg);
            pf[0] = r0[0]; pf[1] = r0[1]; pf[2] = r0[2]; pf[3] = r0[3];
            pf[4] = r1[0]; pf[5] = r1[1]; pf[6] = r1[2]; pf[7] = r1[3];
        }

        // ---- Phase B: QK WMMA, one 16x16 score tile per wave ----
        {
            const int st = qq;
            v8f c;
            #pragma unroll
            for (int r = 0; r < 8; ++r) c[r] = 0.f;
            #pragma unroll
            for (int kk = 0; kk < 4; ++kk) {
                BFPack bop;   // B = K^T: lane -> key row, d in table order
                #pragma unroll
                for (int v = 0; v < 8; ++v)
                    bop.u[v] = Xrow32[(st * 16 + lm) * 64 + 16 * kk + (v < 4 ? 4 * h + v : 8 + 4 * h + (v - 4))];
                c = __builtin_amdgcn_wmma_f32_16x16x32_bf16(
                        false, qa[kk].v, false, bop.v, (short)0, c, false, false);
            }
            float bias = Mbias[s0 + st * 16 + lm];
            #pragma unroll
            for (int r = 0; r < 8; ++r) {
                int m = mt * 16 + r + 8 * h;                 // C layout: M = r + 8h
                Sbuf[m * SC + st * 16 + lm] = c[r] * scale + bias;
            }
        }
        __syncthreads();

        // ---- Phase C/D: online softmax (scores kept in registers) ----
        {
            int row = tid >> 3, seg = tid & 7;
            float sv[8];
            float mloc = NEG_INF;
            #pragma unroll
            for (int j = 0; j < 8; ++j) {
                sv[j] = Sbuf[row * SC + seg * 8 + j];
                mloc = fmaxf(mloc, sv[j]);
            }
            redbuf[tid] = mloc;
            __syncthreads();

            // every thread of the row reduces redundantly (no extra barrier)
            float mc = NEG_INF;
            #pragma unroll
            for (int j = 0; j < 8; ++j) mc = fmaxf(mc, redbuf[(row << 3) + j]);
            float mprev = m_state[row];
            float mnew  = fmaxf(mprev, mc);
            float rsc   = (mprev == NEG_INF) ? 0.f : __expf(mprev - mnew);

            float psum = 0.f, p[8];
            #pragma unroll
            for (int j = 0; j < 8; ++j) {
                float pv = (mnew == NEG_INF) ? 0.f : __expf(sv[j] - mnew);
                p[j] = pv; psum += pv;
            }
            #pragma unroll
            for (int j = 0; j < 4; ++j)
                Pbuf32[row * (SC / 2) + seg * 4 + j] = bf16_2(p[2*j], p[2*j+1]);
            redbuf2[tid] = psum;
            __syncthreads();

            if (seg == 0) {
                float lc = 0.f;
                #pragma unroll
                for (int j = 0; j < 8; ++j) lc += redbuf2[(row << 3) + j];
                l_state[row]  = l_state[row] * rsc + lc;
                m_state[row]  = mnew;
                rescale[row]  = rsc;
            }
        }
        __syncthreads();

        // ---- Phase E: rescale accumulators, AV WMMA (2 d-tiles per wave) ----
        {
            float f[8];
            #pragma unroll
            for (int r = 0; r < 8; ++r) f[r] = rescale[mt * 16 + r + 8 * h];
            #pragma unroll
            for (int r = 0; r < 8; ++r) { accO[0][r] *= f[r]; accO[1][r] *= f[r]; }
            #pragma unroll
            for (int kk = 0; kk < SC / 32; ++kk) {
                BFPack aop;  // A = P tile, 16-bit A layout
                #pragma unroll
                for (int v = 0; v < 8; ++v) {
                    int idx = 16 * kk + (v < 4 ? 4 * h + v : 8 + 4 * h + (v - 4));
                    aop.u[v] = Pbuf32[(mt * 16 + lm) * (SC / 2) + idx];
                }
                #pragma unroll
                for (int i = 0; i < 2; ++i) {
                    int dt = qq + 4 * i;
                    BFPack bop;  // B = V: lane -> output column d, s in table order
                    #pragma unroll
                    for (int v = 0; v < 8; ++v)
                        bop.u[v] = Xt32[(dt * 16 + lm) * (SC / 2) + 16 * kk + (v < 4 ? 4 * h + v : 8 + 4 * h + (v - 4))];
                    accO[i] = __builtin_amdgcn_wmma_f32_16x16x32_bf16(
                                  false, aop.v, false, bop.v, (short)0, accO[i], false, false);
                }
            }
        }
        __syncthreads();
    }

    // ---- Epilogue: normalize by row sums, store real rows only ----
    #pragma unroll
    for (int i = 0; i < 2; ++i) {
        int dt = qq + 4 * i;
        #pragma unroll
        for (int r = 0; r < 8; ++r) {
            int m = mt * 16 + r + 8 * h;
            if (m < NQ) {
                float lv = l_state[m];
                float o  = accO[i][r] * (lv > 0.f ? 1.f / lv : 0.f);
                out[((size_t)b * NQ + m) * DD + dt * 16 + lm] = o;
            }
        }
    }
}

extern "C" void kernel_launch(void* const* d_in, const int* in_sizes, int n_in,
                              void* d_out, int out_size, void* d_ws, size_t ws_size,
                              hipStream_t stream) {
    (void)in_sizes; (void)n_in; (void)out_size; (void)d_ws; (void)ws_size;
    const float* x         = (const float*)d_in[0];
    const float* q_emb     = (const float*)d_in[1];
    const int*   questions = (const int*)d_in[2];
    const int*   mask      = (const int*)d_in[3];
    float*       out       = (float*)d_out;
    attn_pool_kernel<<<dim3(BB), dim3(256), 0, stream>>>(x, q_emb, questions, mask, out);
}
